// CustomModel_11536282157205
// MI455X (gfx1250) — compile-verified, matrix operand order checked by reading, not run on hardware
//
#include <hip/hip_runtime.h>
#include <hip/hip_bf16.h>
#include <math.h>

// Problem constants (from reference): B=8, N=128, H=768, C=5
#define BDIM 8
#define NDIM 128
#define HDIM 768
#define CDIM 5
#define PDIM 8256   // N*(N+1)/2
#define NT64 12     // HDIM / 64  (column super-tiles per row)

typedef __attribute__((ext_vector_type(16))) __bf16        v16bf;
typedef __attribute__((ext_vector_type(8)))  float         v8f;
typedef __attribute__((ext_vector_type(4)))  float         f32x4;
typedef __attribute__((ext_vector_type(4)))  unsigned int  u32x4;

// Load a 16-element bf16 WMMA fragment half-row: elements [0..7] from p[0..7],
// elements [8..15] from p[16..23] (per ISA 7.12.2 16-bit A/B layout, wave32).
static __device__ inline v16bf load_bf16_frag(const __bf16* p) {
  union { v16bf v; u32x4 q[2]; } u;
  u.q[0] = *reinterpret_cast<const u32x4*>(p);
  u.q[1] = *reinterpret_cast<const u32x4*>(p + 16);
  return u.v;
}

// ---------------------------------------------------------------------------
// K0: fold weights (bf16, transposed for B-fragment reads), bf16 h, pair table
// ---------------------------------------------------------------------------
__global__ __launch_bounds__(256) void prep_kernel(
    const float* __restrict__ h, const float* __restrict__ Wcat,
    __bf16* __restrict__ hb, __bf16* __restrict__ WaT,
    __bf16* __restrict__ WbT, __bf16* __restrict__ W4T,
    int* __restrict__ pi, int* __restrict__ pj) {
  const int t = blockIdx.x * blockDim.x + threadIdx.x;
  if (t < HDIM * HDIM) {
    const int hin = t / HDIM;        // input dim (K of the GEMMs)
    const int k   = t % HDIM;        // output dim (N of the GEMMs)
    const float w1 = Wcat[(size_t)hin * HDIM + k];
    const float w2 = Wcat[(size_t)(HDIM + hin) * HDIM + k];
    const float w3 = Wcat[(size_t)(2 * HDIM + hin) * HDIM + k];
    const float w4 = Wcat[(size_t)(3 * HDIM + hin) * HDIM + k];
    const size_t tp = (size_t)k * HDIM + hin;   // transposed: [out][in]
    WaT[tp] = (__bf16)(w1 + w3);
    WbT[tp] = (__bf16)(w2 - w3);
    W4T[tp] = (__bf16)w4;
  }
  if (t < BDIM * NDIM * HDIM) hb[t] = (__bf16)h[t];
  if (t < NDIM) {
    const int i = t;
    const int base = i * NDIM - (i * (i - 1)) / 2;   // triu row offset
    for (int j = i; j < NDIM; ++j) { pi[base + j - i] = i; pj[base + j - i] = j; }
  }
}

// ---------------------------------------------------------------------------
// K1: A = h @ (W1+W3), Bm = h @ (W2-W3)  via bf16 WMMA, fp32 accumulate.
// One wave computes a 16x64 tile; K-loop manually double-buffered so the
// next step's 5 fragment loads overlap the current step's 4 WMMAs.
// ---------------------------------------------------------------------------
__global__ __launch_bounds__(32) void gemm_ab_kernel(
    const __bf16* __restrict__ hb, const __bf16* __restrict__ WaT,
    const __bf16* __restrict__ WbT, float* __restrict__ Abuf,
    float* __restrict__ Bbuf) {
  const int m0 = blockIdx.x * 16;              // row in flattened [B*N]
  const int n0 = blockIdx.y * 64;              // output feature super-tile
  const __bf16* WT = blockIdx.z ? WbT : WaT;
  float* outp      = blockIdx.z ? Bbuf : Abuf;
  const int lane = threadIdx.x;
  const int col  = lane & 15;
  const int hf   = lane >> 4;
  const __bf16* aRow = hb + (size_t)(m0 + col) * HDIM;
  const __bf16* bRow = WT + (size_t)(n0 + col) * HDIM;
  v8f acc[4] = {};

  // prologue: fragments for K-step 0
  int base = hf * 8;
  v16bf a = load_bf16_frag(aRow + base);
  v16bf b0 = load_bf16_frag(bRow + base);
  v16bf b1 = load_bf16_frag(bRow + (size_t)16 * HDIM + base);
  v16bf b2 = load_bf16_frag(bRow + (size_t)32 * HDIM + base);
  v16bf b3 = load_bf16_frag(bRow + (size_t)48 * HDIM + base);

  for (int k0 = 32; k0 < HDIM; k0 += 32) {
    const int nb = k0 + hf * 8;
    // issue next-step loads before consuming current fragments
    v16bf na  = load_bf16_frag(aRow + nb);
    v16bf nb0 = load_bf16_frag(bRow + nb);
    v16bf nb1 = load_bf16_frag(bRow + (size_t)16 * HDIM + nb);
    v16bf nb2 = load_bf16_frag(bRow + (size_t)32 * HDIM + nb);
    v16bf nb3 = load_bf16_frag(bRow + (size_t)48 * HDIM + nb);
    acc[0] = __builtin_amdgcn_wmma_f32_16x16x32_bf16(false, a, false, b0,
                                                     (short)0, acc[0], false, false);
    acc[1] = __builtin_amdgcn_wmma_f32_16x16x32_bf16(false, a, false, b1,
                                                     (short)0, acc[1], false, false);
    acc[2] = __builtin_amdgcn_wmma_f32_16x16x32_bf16(false, a, false, b2,
                                                     (short)0, acc[2], false, false);
    acc[3] = __builtin_amdgcn_wmma_f32_16x16x32_bf16(false, a, false, b3,
                                                     (short)0, acc[3], false, false);
    a = na; b0 = nb0; b1 = nb1; b2 = nb2; b3 = nb3;
  }
  // epilogue: last K-step
  acc[0] = __builtin_amdgcn_wmma_f32_16x16x32_bf16(false, a, false, b0,
                                                   (short)0, acc[0], false, false);
  acc[1] = __builtin_amdgcn_wmma_f32_16x16x32_bf16(false, a, false, b1,
                                                   (short)0, acc[1], false, false);
  acc[2] = __builtin_amdgcn_wmma_f32_16x16x32_bf16(false, a, false, b2,
                                                   (short)0, acc[2], false, false);
  acc[3] = __builtin_amdgcn_wmma_f32_16x16x32_bf16(false, a, false, b3,
                                                   (short)0, acc[3], false, false);

#pragma unroll
  for (int v = 0; v < 8; ++v) {
    float* orow = outp + (size_t)(m0 + v + hf * 8) * HDIM;
#pragma unroll
    for (int t = 0; t < 4; ++t) orow[n0 + 16 * t + col] = acc[t][v];
  }
}

// ---------------------------------------------------------------------------
// K2: spans: D = (h_i*h_j) @ W4, A-frag built on the fly (fp32 loads + bf16
// convert) and reused across a 16x64 tile. K-loop double-buffered: next-step
// raw fp32 + B-frag loads overlap current-step converts (VALU co-executes
// with WMMA) and the 4 WMMAs. Epilogue adds A[i]+Bm[j]+bW, tanh, bf16 store,
// and a shuffle-reduced score partial (span . h_hat) per row.
// ---------------------------------------------------------------------------
__global__ __launch_bounds__(32) void span_kernel(
    const float* __restrict__ h, const __bf16* __restrict__ W4T,
    const float* __restrict__ Abuf, const float* __restrict__ Bbuf,
    const float* __restrict__ bW, const float* __restrict__ hhat,
    const int* __restrict__ pi, const int* __restrict__ pj,
    __bf16* __restrict__ spans, float* __restrict__ spartial) {
  const int p0 = blockIdx.x * 16;
  const int n0 = blockIdx.y * 64;
  const int b  = blockIdx.z;
  const int lane = threadIdx.x;
  const int col  = lane & 15;
  const int hf   = lane >> 4;

  const int pr = p0 + col;                     // A-frag row this lane feeds
  const int ir = pi[pr], jr = pj[pr];
  const float* hi = h + ((size_t)b * NDIM + ir) * HDIM;
  const float* hj = h + ((size_t)b * NDIM + jr) * HDIM;
  const __bf16* bRow = W4T + (size_t)(n0 + col) * HDIM;

  v8f acc[4] = {};

  // prologue: raw A inputs + B fragments for K-step 0
  int base = hf * 8;
  f32x4 x0 = *reinterpret_cast<const f32x4*>(hi + base);
  f32x4 x1 = *reinterpret_cast<const f32x4*>(hi + base + 4);
  f32x4 x2 = *reinterpret_cast<const f32x4*>(hi + base + 16);
  f32x4 x3 = *reinterpret_cast<const f32x4*>(hi + base + 20);
  f32x4 y0 = *reinterpret_cast<const f32x4*>(hj + base);
  f32x4 y1 = *reinterpret_cast<const f32x4*>(hj + base + 4);
  f32x4 y2 = *reinterpret_cast<const f32x4*>(hj + base + 16);
  f32x4 y3 = *reinterpret_cast<const f32x4*>(hj + base + 20);
  v16bf b0 = load_bf16_frag(bRow + base);
  v16bf b1 = load_bf16_frag(bRow + (size_t)16 * HDIM + base);
  v16bf b2 = load_bf16_frag(bRow + (size_t)32 * HDIM + base);
  v16bf b3 = load_bf16_frag(bRow + (size_t)48 * HDIM + base);

  for (int k0 = 32; k0 < HDIM; k0 += 32) {
    const int nb = k0 + hf * 8;
    // issue next-step loads first
    f32x4 nx0 = *reinterpret_cast<const f32x4*>(hi + nb);
    f32x4 nx1 = *reinterpret_cast<const f32x4*>(hi + nb + 4);
    f32x4 nx2 = *reinterpret_cast<const f32x4*>(hi + nb + 16);
    f32x4 nx3 = *reinterpret_cast<const f32x4*>(hi + nb + 20);
    f32x4 ny0 = *reinterpret_cast<const f32x4*>(hj + nb);
    f32x4 ny1 = *reinterpret_cast<const f32x4*>(hj + nb + 4);
    f32x4 ny2 = *reinterpret_cast<const f32x4*>(hj + nb + 16);
    f32x4 ny3 = *reinterpret_cast<const f32x4*>(hj + nb + 20);
    v16bf nb0 = load_bf16_frag(bRow + nb);
    v16bf nb1 = load_bf16_frag(bRow + (size_t)16 * HDIM + nb);
    v16bf nb2 = load_bf16_frag(bRow + (size_t)32 * HDIM + nb);
    v16bf nb3 = load_bf16_frag(bRow + (size_t)48 * HDIM + nb);
    // build current A fragment (prod -> bf16) and run the 4 WMMAs
    v16bf a;
#pragma unroll
    for (int t = 0; t < 4; ++t) {
      a[t]      = (__bf16)(x0[t] * y0[t]);
      a[4 + t]  = (__bf16)(x1[t] * y1[t]);
      a[8 + t]  = (__bf16)(x2[t] * y2[t]);
      a[12 + t] = (__bf16)(x3[t] * y3[t]);
    }
    acc[0] = __builtin_amdgcn_wmma_f32_16x16x32_bf16(false, a, false, b0,
                                                     (short)0, acc[0], false, false);
    acc[1] = __builtin_amdgcn_wmma_f32_16x16x32_bf16(false, a, false, b1,
                                                     (short)0, acc[1], false, false);
    acc[2] = __builtin_amdgcn_wmma_f32_16x16x32_bf16(false, a, false, b2,
                                                     (short)0, acc[2], false, false);
    acc[3] = __builtin_amdgcn_wmma_f32_16x16x32_bf16(false, a, false, b3,
                                                     (short)0, acc[3], false, false);
    x0 = nx0; x1 = nx1; x2 = nx2; x3 = nx3;
    y0 = ny0; y1 = ny1; y2 = ny2; y3 = ny3;
    b0 = nb0; b1 = nb1; b2 = nb2; b3 = nb3;
  }
  // epilogue: last K-step
  {
    v16bf a;
#pragma unroll
    for (int t = 0; t < 4; ++t) {
      a[t]      = (__bf16)(x0[t] * y0[t]);
      a[4 + t]  = (__bf16)(x1[t] * y1[t]);
      a[8 + t]  = (__bf16)(x2[t] * y2[t]);
      a[12 + t] = (__bf16)(x3[t] * y3[t]);
    }
    acc[0] = __builtin_amdgcn_wmma_f32_16x16x32_bf16(false, a, false, b0,
                                                     (short)0, acc[0], false, false);
    acc[1] = __builtin_amdgcn_wmma_f32_16x16x32_bf16(false, a, false, b1,
                                                     (short)0, acc[1], false, false);
    acc[2] = __builtin_amdgcn_wmma_f32_16x16x32_bf16(false, a, false, b2,
                                                     (short)0, acc[2], false, false);
    acc[3] = __builtin_amdgcn_wmma_f32_16x16x32_bf16(false, a, false, b3,
                                                     (short)0, acc[3], false, false);
  }

#pragma unroll
  for (int v = 0; v < 8; ++v) {
    const int p = p0 + v + hf * 8;             // D row (ISA C/D layout)
    const int ii = pi[p], jj = pj[p];
    const float* arow = Abuf + ((size_t)b * NDIM + ii) * HDIM;
    const float* brow = Bbuf + ((size_t)b * NDIM + jj) * HDIM;
    __bf16* srow = spans + ((size_t)b * PDIM + p) * HDIM;
    float rowsum = 0.f;
#pragma unroll
    for (int t = 0; t < 4; ++t) {
      const int n = n0 + 16 * t + col;
      const float pre = acc[t][v] + arow[n] + brow[n] + bW[n];
      const float sp = tanhf(pre);
      srow[n] = (__bf16)sp;
      rowsum += sp * hhat[n];
    }
    // reduce the 16 columns held by this half-wave (masks stay within half)
#pragma unroll
    for (int m = 1; m < 16; m <<= 1) rowsum += __shfl_xor(rowsum, m, 32);
    if (col == 0)
      spartial[((size_t)b * PDIM + p) * NT64 + blockIdx.y] = rowsum;
  }
}

// ---------------------------------------------------------------------------
// K3: per-batch softmax over P span scores (sum the 12 column-tile partials).
// ---------------------------------------------------------------------------
__global__ __launch_bounds__(256) void softmax_kernel(
    const float* __restrict__ spartial, float* __restrict__ s,
    float* __restrict__ alpha) {
  const int b = blockIdx.x;
  const int tid = threadIdx.x;
  __shared__ float red[256];
  float lmax = -3.4e38f;
  for (int p = tid; p < PDIM; p += 256) {
    const float* sp = spartial + ((size_t)b * PDIM + p) * NT64;
    float acc = 0.f;
#pragma unroll
    for (int t = 0; t < NT64; ++t) acc += sp[t];
    s[(size_t)b * PDIM + p] = acc;
    lmax = fmaxf(lmax, acc);
  }
  red[tid] = lmax; __syncthreads();
  for (int st = 128; st > 0; st >>= 1) {
    if (tid < st) red[tid] = fmaxf(red[tid], red[tid + st]);
    __syncthreads();
  }
  const float m = red[0];
  __syncthreads();
  float lz = 0.f;
  for (int p = tid; p < PDIM; p += 256) {
    const float e = expf(s[(size_t)b * PDIM + p] - m);
    alpha[(size_t)b * PDIM + p] = e;
    lz += e;
  }
  red[tid] = lz; __syncthreads();
  for (int st = 128; st > 0; st >>= 1) {
    if (tid < st) red[tid] += red[tid + st];
    __syncthreads();
  }
  const float inv = 1.f / red[0];
  for (int p = tid; p < PDIM; p += 256) alpha[(size_t)b * PDIM + p] *= inv;
}

// ---------------------------------------------------------------------------
// K4: h_tilde[b,k] = sum_p alpha[b,p] * spans[b,p,k]   (coalesced over k)
// ---------------------------------------------------------------------------
__global__ __launch_bounds__(256) void pool_kernel(
    const __bf16* __restrict__ spans, const float* __restrict__ alpha,
    float* __restrict__ htilde) {
  const int b = blockIdx.y;
  const int k = blockIdx.x * 256 + threadIdx.x;
  const __bf16* sp = spans + (size_t)b * PDIM * HDIM + k;
  const float*  al = alpha + (size_t)b * PDIM;
  float acc = 0.f;
  for (int p = 0; p < PDIM; ++p) acc += al[p] * (float)sp[(size_t)p * HDIM];
  htilde[(size_t)b * HDIM + k] = acc;
}

// ---------------------------------------------------------------------------
// K5: logits = h_tilde @ Wout + bout ; out = log_softmax(logits)
// ---------------------------------------------------------------------------
__global__ __launch_bounds__(64) void head_kernel(
    const float* __restrict__ htilde, const float* __restrict__ Wout,
    const float* __restrict__ bout, float* __restrict__ out) {
  __shared__ float sl[BDIM * CDIM];
  const int t = threadIdx.x;
  if (t < BDIM * CDIM) {
    const int b = t / CDIM, c = t % CDIM;
    float acc = bout[c];
    for (int k = 0; k < HDIM; ++k)
      acc += htilde[(size_t)b * HDIM + k] * Wout[(size_t)k * CDIM + c];
    sl[t] = acc;
  }
  __syncthreads();
  if (t < BDIM) {
    float m = sl[t * CDIM];
    for (int c = 1; c < CDIM; ++c) m = fmaxf(m, sl[t * CDIM + c]);
    float z = 0.f;
    for (int c = 0; c < CDIM; ++c) z += expf(sl[t * CDIM + c] - m);
    const float lse = logf(z);
    for (int c = 0; c < CDIM; ++c) out[t * CDIM + c] = sl[t * CDIM + c] - m - lse;
  }
}

// ---------------------------------------------------------------------------
extern "C" void kernel_launch(void* const* d_in, const int* in_sizes, int n_in,
                              void* d_out, int out_size, void* d_ws, size_t ws_size,
                              hipStream_t stream) {
  const float* h    = (const float*)d_in[0];   // [B,N,H]
  const float* Wcat = (const float*)d_in[1];   // [4H,H]
  const float* bW   = (const float*)d_in[2];   // [H]
  const float* hhat = (const float*)d_in[3];   // [H]
  const float* Wout = (const float*)d_in[4];   // [H,C]
  const float* bout = (const float*)d_in[5];   // [C]
  float* out = (float*)d_out;                  // [B,C]

  char* ws = (char*)d_ws;
  auto alloc = [&](size_t bytes) -> char* {
    char* p = ws;
    ws += (bytes + 255) & ~(size_t)255;
    return p;
  };
  __bf16* hb   = (__bf16*)alloc((size_t)BDIM * NDIM * HDIM * 2);
  __bf16* WaT  = (__bf16*)alloc((size_t)HDIM * HDIM * 2);
  __bf16* WbT  = (__bf16*)alloc((size_t)HDIM * HDIM * 2);
  __bf16* W4T  = (__bf16*)alloc((size_t)HDIM * HDIM * 2);
  float*  Abuf = (float*)alloc((size_t)BDIM * NDIM * HDIM * 4);
  float*  Bbuf = (float*)alloc((size_t)BDIM * NDIM * HDIM * 4);
  int*    pi   = (int*)alloc((size_t)PDIM * 4);
  int*    pj   = (int*)alloc((size_t)PDIM * 4);
  __bf16* spans = (__bf16*)alloc((size_t)BDIM * PDIM * HDIM * 2);   // ~101 MB
  float*  spart = (float*)alloc((size_t)BDIM * PDIM * NT64 * 4);    // ~3.2 MB
  float*  s     = (float*)alloc((size_t)BDIM * PDIM * 4);
  float*  alpha = (float*)alloc((size_t)BDIM * PDIM * 4);
  float*  htil  = (float*)alloc((size_t)BDIM * HDIM * 4);

  const int prepThreads = BDIM * NDIM * HDIM;          // covers H*H too
  prep_kernel<<<(prepThreads + 255) / 256, 256, 0, stream>>>(
      h, Wcat, hb, WaT, WbT, W4T, pi, pj);

  gemm_ab_kernel<<<dim3((BDIM * NDIM) / 16, NT64, 2), 32, 0, stream>>>(
      hb, WaT, WbT, Abuf, Bbuf);

  span_kernel<<<dim3(PDIM / 16, NT64, BDIM), 32, 0, stream>>>(
      h, W4T, Abuf, Bbuf, bW, hhat, pi, pj, spans, spart);

  softmax_kernel<<<dim3(BDIM), 256, 0, stream>>>(spart, s, alpha);

  pool_kernel<<<dim3(HDIM / 256, BDIM), 256, 0, stream>>>(spans, alpha, htil);

  head_kernel<<<1, 64, 0, stream>>>(htil, Wout, bout, out);
}
